// NTLayer_53420803227696
// MI455X (gfx1250) — compile-verified, matrix operand order checked by reading, not run on hardware
//
#include <hip/hip_runtime.h>
#include <hip/hip_bf16.h>

typedef __attribute__((ext_vector_type(16))) _Float16 v16h;
typedef __attribute__((ext_vector_type(8)))  float    v8f;

#define NNODE 8192
#define DEG   32
#define NPB   2            // nodes per block
#define MROWS (NPB * DEG)  // 64 edge rows per block
#define DIN   256
#define EDIM  64
#define DD    512          // internal width D
#define KFEAT 576          // 2*DIN + EDIM
#define NH    8
#define HC    64           // channels per head
#define CP    32           // Cp = D/(H*2)

#define FEAT_LD 584
#define QKV_LD  1544
#define EX_LD   520
#define H_LD    520
#define SC_LD   36

// Branch-free tanh-form GELU (avoids erff's divergent polynomial blowup).
__device__ __forceinline__ float gelu_fast(float x) {
  float u = 0.7978845608028654f * x * (1.0f + 0.044715f * x * x);
  float t = __expf(-2.0f * fabsf(u));                 // e^{-2|u|}
  float th = (1.0f - t) * __builtin_amdgcn_rcpf(1.0f + t);  // tanh(|u|)
  th = copysignf(th, u);
  return 0.5f * x * (1.0f + th);
}

// A or B fragment for v_wmma_f32_16x16x32_f16 from a row-major f16 source.
// A: rows are M (16), k over K-chunk of 32.  B: rows are the output column
// index (B[k][n] = W[n][k] for X @ W^T).  Per ISA layout: lanes 0-15 hold
// row (lane&15) with K 0-7 / 16-23; lanes 16-31 hold K 8-15 / 24-31.
__device__ __forceinline__ v16h frag_rowmajor(const _Float16* base, int ld, int row0, int k0) {
  const int lane = threadIdx.x & 31;
  const _Float16* p = base + (row0 + (lane & 15)) * ld + k0 + ((lane >> 4) << 3);
  v16h f;
#pragma unroll
  for (int e = 0; e < 8; ++e) f[e] = p[e];
#pragma unroll
  for (int e = 0; e < 8; ++e) f[e + 8] = p[e + 16];
  return f;
}

// B fragment when the source is row-major [K][N] (B[k][n] read down a column).
__device__ __forceinline__ v16h frag_colmajor(const _Float16* base, int ld, int k0, int n0) {
  const int lane = threadIdx.x & 31;
  const int kh = (lane >> 4) << 3;
  const _Float16* p = base + n0 + (lane & 15);
  v16h f;
#pragma unroll
  for (int e = 0; e < 8; ++e) f[e] = p[(k0 + kh + e) * ld];
#pragma unroll
  for (int e = 0; e < 8; ++e) f[e + 8] = p[(k0 + kh + 16 + e) * ld];
  return f;
}

__device__ __forceinline__ v8f wmma16(v16h a, v16h b, v8f c) {
  return __builtin_amdgcn_wmma_f32_16x16x32_f16(false, a, false, b, (short)0, c, false, false);
}

__global__ void cvt_f32_f16_kernel(const float* __restrict__ src, _Float16* __restrict__ dst, int n) {
  int i = blockIdx.x * blockDim.x + threadIdx.x;
  if (i < n) dst[i] = (_Float16)src[i];
}

__global__ void zero_f32_kernel(float* __restrict__ dst, int n) {
  int i = blockIdx.x * blockDim.x + threadIdx.x;
  if (i < n) dst[i] = 0.0f;
}

// One block per NPB=2 nodes (64 edges), 256 threads = 8 waves.
// M=64 => every weight (B) fragment feeds 4 M-tiles, halving L2 weight traffic.
__global__ __launch_bounds__(256)
void nt_node_fused_kernel(const float* __restrict__ x,
                          const float* __restrict__ edge_attr,
                          const int*   __restrict__ e,
                          const float* __restrict__ b_comb,
                          const float* __restrict__ b_in,
                          const float* __restrict__ b_out,
                          const _Float16* __restrict__ Wc16,   // [512][576]
                          const _Float16* __restrict__ Wi16,   // [1536][512]
                          const _Float16* __restrict__ Wo16,   // [512][512]
                          float* __restrict__ out) {
  __shared__ alignas(16) union {
    _Float16 feat[MROWS * FEAT_LD];   // gathered edge features (f16)
    _Float16 qkv[MROWS * QKV_LD];     // QKV; q-region reused for attention o
  } sA;
  __shared__ alignas(16) union {
    _Float16 ex[MROWS * EX_LD];             // comb output
    _Float16 sc[NPB * NH * DEG * SC_LD];    // attention weights
    _Float16 h[MROWS * H_LD];               // post out-proj gelu
  } sB;
  __shared__ float s_logit[MROWS * NH];
  __shared__ float s_alpha[MROWS * NH];
  __shared__ int   s_e1[MROWS];

  const int blk  = blockIdx.x;
  const int tid  = threadIdx.x;
  const int wave = tid >> 5;
  const int lane = tid & 31;
  const int hi   = lane >> 4;
  const int ln   = lane & 15;
  const int m_total = NNODE * DEG;
  const int edge0 = blk * MROWS;

  if (tid < MROWS) s_e1[tid] = e[m_total + edge0 + tid];
  __syncthreads();

  // ---- stage feat[j] = [ x[src] | x[e1[j]] | edge_attr[j] ] as f16 ----
  {
    const int row = tid >> 2;   // edge row 0..63
    const int seg = tid & 3;    // 4 threads per row, 144 cols each
    const float* xsrc  = x + (size_t)(blk * NPB + (row >> 5)) * DIN;
    const float* xdst  = x + (size_t)s_e1[row] * DIN;
    const float* easrc = edge_attr + (size_t)(edge0 + row) * EDIM;
    _Float16* frow = sA.feat + row * FEAT_LD;
    for (int i = 0; i < 144; ++i) {
      int col = seg * 144 + i;
      float v;
      if (col < DIN)          v = xsrc[col];
      else if (col < 2 * DIN) v = xdst[col - DIN];
      else                    v = easrc[col - 2 * DIN];
      frow[col] = (_Float16)v;
    }
  }
  __syncthreads();

  // ---- GEMM1: ex = gelu(gelu(feat @ Wcomb^T + b)) ; [64,576]x[576,512] ----
  {
    const int nbase = wave * 64;
    v8f acc[4][4] = {};
    for (int kk = 0; kk < KFEAT; kk += 32) {
      v16h a[4];
#pragma unroll
      for (int mt = 0; mt < 4; ++mt) a[mt] = frag_rowmajor(sA.feat, FEAT_LD, mt * 16, kk);
#pragma unroll
      for (int nt = 0; nt < 4; ++nt) {
        v16h b = frag_rowmajor(Wc16, KFEAT, nbase + nt * 16, kk);
#pragma unroll
        for (int mt = 0; mt < 4; ++mt) acc[mt][nt] = wmma16(a[mt], b, acc[mt][nt]);
      }
    }
#pragma unroll
    for (int mt = 0; mt < 4; ++mt)
#pragma unroll
      for (int nt = 0; nt < 4; ++nt) {
        int col = nbase + nt * 16 + ln;
        float bias = b_comb[col];
#pragma unroll
        for (int r = 0; r < 8; ++r) {
          int row = mt * 16 + hi * 8 + r;
          sB.ex[row * EX_LD + col] = (_Float16)gelu_fast(gelu_fast(acc[mt][nt][r] + bias));
        }
      }
  }
  __syncthreads();

  // ---- GEMM2: qkv = ex @ Win^T + b_in ; each wave computes its head ----
  {
    const int head = wave;
    for (int segq = 0; segq < 3; ++segq) {
      const int colbase = segq * DD + head * HC;
      v8f acc[4][4] = {};
      for (int kk = 0; kk < DD; kk += 32) {
        v16h a[4];
#pragma unroll
        for (int mt = 0; mt < 4; ++mt) a[mt] = frag_rowmajor(sB.ex, EX_LD, mt * 16, kk);
#pragma unroll
        for (int nt = 0; nt < 4; ++nt) {
          v16h b = frag_rowmajor(Wi16, DD, colbase + nt * 16, kk);
#pragma unroll
          for (int mt = 0; mt < 4; ++mt) acc[mt][nt] = wmma16(a[mt], b, acc[mt][nt]);
        }
      }
#pragma unroll
      for (int mt = 0; mt < 4; ++mt)
#pragma unroll
        for (int nt = 0; nt < 4; ++nt) {
          int col = colbase + nt * 16 + ln;
          float bias = b_in[col];
#pragma unroll
          for (int r = 0; r < 8; ++r) {
            int row = mt * 16 + hi * 8 + r;
            sA.qkv[row * QKV_LD + col] = (_Float16)(acc[mt][nt][r] + bias);
          }
        }
    }
  }
  __syncthreads();

  // ---- Attention: one wave per head, loop over the NPB nodes ----
  {
    const int head = wave;
    for (int nd = 0; nd < NPB; ++nd) {
      const int rb = nd * DEG;
      const _Float16* qbase = sA.qkv + head * HC;
      const _Float16* kbase = sA.qkv + DD + head * HC;
      const _Float16* vbase = sA.qkv + (size_t)rb * QKV_LD + 2 * DD + head * HC;
      _Float16* sch = sB.sc + (nd * NH + head) * DEG * SC_LD;

      v8f s[2][2] = {};
      for (int kk = 0; kk < HC; kk += 32) {
        v16h a0 = frag_rowmajor(qbase, QKV_LD, rb, kk);
        v16h a1 = frag_rowmajor(qbase, QKV_LD, rb + 16, kk);
#pragma unroll
        for (int nt = 0; nt < 2; ++nt) {
          v16h b = frag_rowmajor(kbase, QKV_LD, rb + nt * 16, kk);  // B[c][n]=K[n][c]
          s[0][nt] = wmma16(a0, b, s[0][nt]);
          s[1][nt] = wmma16(a1, b, s[1][nt]);
        }
      }
      // Row softmax: row lives in one 16-lane half; xor masks 1..8 stay in-half.
      const float scale = 0.125f;  // 1/sqrt(64)
#pragma unroll
      for (int mt = 0; mt < 2; ++mt) {
#pragma unroll
        for (int r = 0; r < 8; ++r) {
          float a0 = s[mt][0][r] * scale;
          float a1 = s[mt][1][r] * scale;
          float mx = fmaxf(a0, a1);
#pragma unroll
          for (int msk = 1; msk < 16; msk <<= 1) mx = fmaxf(mx, __shfl_xor(mx, msk, 32));
          float e0 = __expf(a0 - mx);
          float e1v = __expf(a1 - mx);
          float sm = e0 + e1v;
#pragma unroll
          for (int msk = 1; msk < 16; msk <<= 1) sm += __shfl_xor(sm, msk, 32);
          float inv = __builtin_amdgcn_rcpf(sm);
          int row = mt * 16 + hi * 8 + r;        // row within node
          sch[row * SC_LD + ln]      = (_Float16)(e0 * inv);
          sch[row * SC_LD + 16 + ln] = (_Float16)(e1v * inv);
        }
      }
      // o = attn[32,32] @ v[32,64]; write o over the (dead) q region
      v8f o[2][4] = {};
      {
        v16h a0 = frag_rowmajor(sch, SC_LD, 0, 0);
        v16h a1 = frag_rowmajor(sch, SC_LD, 16, 0);
#pragma unroll
        for (int nt = 0; nt < 4; ++nt) {
          v16h b = frag_colmajor(vbase, QKV_LD, 0, nt * 16);  // B[k][n]=V[k][n]
          o[0][nt] = wmma16(a0, b, o[0][nt]);
          o[1][nt] = wmma16(a1, b, o[1][nt]);
        }
      }
#pragma unroll
      for (int mt = 0; mt < 2; ++mt)
#pragma unroll
        for (int nt = 0; nt < 4; ++nt) {
          int col = head * HC + nt * 16 + ln;
#pragma unroll
          for (int r = 0; r < 8; ++r) {
            int row = rb + mt * 16 + hi * 8 + r;
            sA.qkv[row * QKV_LD + col] = (_Float16)o[mt][nt][r];
          }
        }
    }
  }
  __syncthreads();

  // ---- GEMM3: h = gelu(o @ Wout^T + b_out) ----
  {
    const int nbase = wave * 64;
    v8f acc[4][4] = {};
    for (int kk = 0; kk < DD; kk += 32) {
      v16h a[4];
#pragma unroll
      for (int mt = 0; mt < 4; ++mt) a[mt] = frag_rowmajor(sA.qkv, QKV_LD, mt * 16, kk);
#pragma unroll
      for (int nt = 0; nt < 4; ++nt) {
        v16h b = frag_rowmajor(Wo16, DD, nbase + nt * 16, kk);
#pragma unroll
        for (int mt = 0; mt < 4; ++mt) acc[mt][nt] = wmma16(a[mt], b, acc[mt][nt]);
      }
    }
#pragma unroll
    for (int mt = 0; mt < 4; ++mt)
#pragma unroll
      for (int nt = 0; nt < 4; ++nt) {
        int col = nbase + nt * 16 + ln;
        float bias = b_out[col];
#pragma unroll
        for (int r = 0; r < 8; ++r) {
          int row = mt * 16 + hi * 8 + r;
          sB.h[row * H_LD + col] = (_Float16)gelu_fast(acc[mt][nt][r] + bias);
        }
      }
  }
  __syncthreads();

  // ---- gate logits = mean of channel-1 half per head ----
  for (int t = tid; t < MROWS * NH; t += 256) {
    const int row = t >> 3;
    const int head = t & 7;
    const _Float16* hrow = sB.h + row * H_LD + head * HC + CP;
    float sum = 0.0f;
#pragma unroll
    for (int c = 0; c < CP; ++c) sum += (float)hrow[c];
    s_logit[row * NH + head] = sum * (1.0f / CP);
  }
  __syncthreads();
  // ---- softmax over the 32 edges of each node, per head ----
  if (tid < NPB * NH) {
    const int nd = tid >> 3;
    const int head = tid & 7;
    const int rb = nd * DEG;
    float mx = -1e30f;
    for (int j = 0; j < DEG; ++j) mx = fmaxf(mx, s_logit[(rb + j) * NH + head]);
    float sm = 0.0f;
    for (int j = 0; j < DEG; ++j) {
      float ev = __expf(s_logit[(rb + j) * NH + head] - mx);
      s_alpha[(rb + j) * NH + head] = ev;
      sm += ev;
    }
    float inv = 1.0f / sm;
    for (int j = 0; j < DEG; ++j) s_alpha[(rb + j) * NH + head] *= inv;
  }
  __syncthreads();
  // ---- wz = h[:, :, 0] * alpha ; scatter-add to out[e1] ----
  {
    const int head = tid >> 5;
    const int cp = tid & 31;
    for (int j = 0; j < MROWS; ++j) {
      float v = (float)sB.h[j * H_LD + head * HC + cp] * s_alpha[j * NH + head];
      atomicAdd(out + (size_t)s_e1[j] * (NH * CP) + head * CP + cp, v);
    }
  }
}

extern "C" void kernel_launch(void* const* d_in, const int* in_sizes, int n_in,
                              void* d_out, int out_size, void* d_ws, size_t ws_size,
                              hipStream_t stream) {
  const float* x   = (const float*)d_in[0];
  const float* ea  = (const float*)d_in[1];
  const int*   e   = (const int*)d_in[2];
  const float* Wc  = (const float*)d_in[3];
  const float* bc  = (const float*)d_in[4];
  const float* Wi  = (const float*)d_in[5];
  const float* bi  = (const float*)d_in[6];
  const float* Wo  = (const float*)d_in[7];
  const float* bo  = (const float*)d_in[8];
  float* out = (float*)d_out;

  // Workspace: f16 weight copies (~2.7 MB), L2 resident.
  _Float16* wc16 = (_Float16*)d_ws;
  _Float16* wi16 = wc16 + (size_t)DD * KFEAT;
  _Float16* wo16 = wi16 + (size_t)(3 * DD) * DD;

  const int ncvt_c = DD * KFEAT;        // 294912
  const int ncvt_i = 3 * DD * DD;       // 786432
  const int ncvt_o = DD * DD;           // 262144
  cvt_f32_f16_kernel<<<(ncvt_c + 255) / 256, 256, 0, stream>>>(Wc, wc16, ncvt_c);
  cvt_f32_f16_kernel<<<(ncvt_i + 255) / 256, 256, 0, stream>>>(Wi, wi16, ncvt_i);
  cvt_f32_f16_kernel<<<(ncvt_o + 255) / 256, 256, 0, stream>>>(Wo, wo16, ncvt_o);
  zero_f32_kernel<<<(out_size + 255) / 256, 256, 0, stream>>>(out, out_size);

  nt_node_fused_kernel<<<NNODE / NPB, 256, 0, stream>>>(x, ea, e, bc, bi, bo,
                                                        wc16, wi16, wo16, out);
}
